// LinearAttentionWrapper_19275813224538
// MI455X (gfx1250) — matrix operand order repaired and checked by reading, not compiled, not run on hardware
//
#include <hip/hip_runtime.h>
#include <hip/hip_bf16.h>
#include <math.h>
#include <stdint.h>

typedef __attribute__((ext_vector_type(16))) _Float16 v16h;
typedef __attribute__((ext_vector_type(8)))  _Float16 v8h;
typedef __attribute__((ext_vector_type(4)))  _Float16 v4h;
typedef __attribute__((ext_vector_type(8)))  float    v8f;

// Problem constants (B=4, T=2048, D=1024, H=16, HD=64, F=128, C=64)
#define B_  4
#define T_  2048
#define D_  1024
#define H_  16
#define HD_ 64
#define F_  128
#define C_  64
#define NCHUNK_ 32

__device__ __forceinline__ int laneid() { return threadIdx.x & 31; }

__device__ __forceinline__ v8f wmma_f16(v16h a, v16h b, v8f c) {
  return __builtin_amdgcn_wmma_f32_16x16x32_f16(false, a, false, b, (short)0, c,
                                                false, false);
}

// ---- CDNA5 async global->LDS copy (ASYNCcnt), 16B per lane ----
__device__ __forceinline__ void async_b128(const _Float16* g, _Float16* l) {
  uint32_t loff = (uint32_t)(uintptr_t)l;   // low 32 bits of generic = LDS offset
  asm volatile("global_load_async_to_lds_b128 %0, %1, off"
               :: "v"(loff), "v"(g) : "memory");
}
__device__ __forceinline__ void wait_async0() {
  asm volatile("s_wait_asynccnt 0x0" ::: "memory");
}

// ---- WMMA fragment helpers (wave32; A/B use contiguous k = 16*(lane/16) + i) ----
// All hot-loop fragment loads are lane-contiguous -> ds_load_b128.

// A fragment 16x32 f16; src row-major [m][k].
__device__ __forceinline__ v16h fragA(const _Float16* p, int m0, int k0, int ld) {
  const int ln = laneid();
  const _Float16* pr = p + (size_t)(m0 + (ln & 15)) * ld + k0 + ((ln >> 4) << 4);
  v16h a;
#pragma unroll
  for (int i = 0; i < 16; ++i) a[i] = pr[i];
  return a;
}

// B fragment 32x16 f16 where B(k,n) = p[n*ld + k] (n-major / transposed store).
__device__ __forceinline__ v16h fragBT(const _Float16* p, int k0, int n0, int ld) {
  const int ln = laneid();
  const _Float16* pr = p + (size_t)(n0 + (ln & 15)) * ld + k0 + ((ln >> 4) << 4);
  v16h b;
#pragma unroll
  for (int i = 0; i < 16; ++i) b[i] = pr[i];
  return b;
}

// B fragment from fp32 n-major source: B(k,n) = (f16)p[n*ld + k].
__device__ __forceinline__ v16h fragBTf32(const float* p, int k0, int n0, int ld) {
  const int ln = laneid();
  const float* pr = p + (size_t)(n0 + (ln & 15)) * ld + k0 + ((ln >> 4) << 4);
  v16h b;
#pragma unroll
  for (int i = 0; i < 16; ++i) b[i] = (_Float16)pr[i];
  return b;
}

// C/D fragment helpers, standard orientation (row-major dest).
__device__ __forceinline__ void storeC(float* p, v8f c, int m0, int n0, int ld) {
  const int ln = laneid();
  float* pr = p + (size_t)(m0 + ((ln >> 4) << 3)) * ld + n0 + (ln & 15);
#pragma unroll
  for (int j = 0; j < 8; ++j) pr[(size_t)j * ld] = c[j];
}
__device__ __forceinline__ void storeCh(_Float16* p, v8f c, int m0, int n0, int ld) {
  const int ln = laneid();
  _Float16* pr = p + (size_t)(m0 + ((ln >> 4) << 3)) * ld + n0 + (ln & 15);
#pragma unroll
  for (int j = 0; j < 8; ++j) pr[(size_t)j * ld] = (_Float16)c[j];
}
// C/D fragment on a transposed (n-major) fp32 buffer: element (m,n) at p[n*ld+m].
// Contiguous in m -> b128-mergeable.
__device__ __forceinline__ v8f loadCT(const float* p, int m0, int n0, int ld) {
  const int ln = laneid();
  const float* pr = p + (size_t)(n0 + (ln & 15)) * ld + m0 + ((ln >> 4) << 3);
  v8f c;
#pragma unroll
  for (int j = 0; j < 8; ++j) c[j] = pr[j];
  return c;
}
__device__ __forceinline__ void storeCT(float* p, v8f c, int m0, int n0, int ld) {
  const int ln = laneid();
  float* pr = p + (size_t)(n0 + (ln & 15)) * ld + m0 + ((ln >> 4) << 3);
#pragma unroll
  for (int j = 0; j < 8; ++j) pr[j] = c[j];
}

// ------------------------------------------------------------------
// K0: resid = h + r (fp32 out), x = rmsnorm(resid)*w -> f16
// ------------------------------------------------------------------
__global__ void k_addnorm(const float* __restrict__ h, const float* __restrict__ r,
                          const float* __restrict__ w, float* __restrict__ resid,
                          _Float16* __restrict__ xh) {
  const int row = blockIdx.x;
  const int tid = threadIdx.x;
  const size_t base = (size_t)row * D_ + tid * 4;
  float4 a = *(const float4*)(h + base);
  float4 b = *(const float4*)(r + base);
  float4 s = {a.x + b.x, a.y + b.y, a.z + b.z, a.w + b.w};
  *(float4*)(resid + base) = s;
  float sq = s.x * s.x + s.y * s.y + s.z * s.z + s.w * s.w;
#pragma unroll
  for (int off = 16; off > 0; off >>= 1) sq += __shfl_xor(sq, off);
  __shared__ float red[8];
  if ((tid & 31) == 0) red[tid >> 5] = sq;
  __syncthreads();
  float tot = 0.f;
#pragma unroll
  for (int i = 0; i < 8; ++i) tot += red[i];
  const float inv = rsqrtf(tot * (1.0f / D_) + 1e-5f);
  float4 wv = *(const float4*)(w + tid * 4);
  v4h o;
  o[0] = (_Float16)(s.x * inv * wv.x);
  o[1] = (_Float16)(s.y * inv * wv.y);
  o[2] = (_Float16)(s.z * inv * wv.z);
  o[3] = (_Float16)(s.w * inv * wv.w);
  *(v4h*)(xh + base) = o;
}

// ------------------------------------------------------------------
// K1a: fold hedgehog linear into projection weight:
//   out[i, h*64+e] = (f16) sum_d w[i, h*64+d] * hh[e, d]
// ------------------------------------------------------------------
__global__ void k_fold(const float* __restrict__ w, const float* __restrict__ hh,
                       _Float16* __restrict__ out) {
  const int gid  = blockIdx.x * 256 + threadIdx.x;   // 0 .. D*D-1
  const int i    = gid >> 10;
  const int col  = gid & (D_ - 1);
  const int head = col >> 6;
  const int e    = col & 63;
  const float* wr = w + (size_t)i * D_ + (head << 6);
  const float* hr = hh + e * HD_;
  float s = 0.f;
#pragma unroll 8
  for (int d = 0; d < HD_; ++d) s += wr[d] * hr[d];
  out[gid] = (_Float16)s;
}

__global__ void k_cvt(const float* __restrict__ in, _Float16* __restrict__ out) {
  const int gid = blockIdx.x * 256 + threadIdx.x;
  out[gid] = (_Float16)in[gid];
}

// ------------------------------------------------------------------
// K2: WMMA GEMM  C[M,N] = A[M,K] @ B[K,N], f16 in, f32 accum.
// 128x128 block tile, 8 waves of 64x32.  B tile staged TRANSPOSED in LDS so
// both A and B fragments load as contiguous ds_load_b128.
// ------------------------------------------------------------------
__global__ void k_gemm(const _Float16* __restrict__ A, const _Float16* __restrict__ Bm,
                       float* __restrict__ Cf, _Float16* __restrict__ Ch,
                       int M, int N, int K, int half_out) {
  constexpr int LDA  = 40;   // k-extent 32 + pad
  constexpr int LDBT = 40;   // transposed B: [n][k], k-extent 32 + pad
  __shared__ _Float16 As [128 * LDA];
  __shared__ _Float16 BsT[128 * LDBT];
  const int tid = threadIdx.x;
  const int wid = tid >> 5;
  const int m0 = blockIdx.y * 128;
  const int n0 = blockIdx.x * 128;
  const int arow = tid >> 1, aseg = (tid & 1) << 4;   // 128 rows x 32
  const int brow = tid >> 3, bseg = (tid & 7) << 4;   // 32 k-rows x 128 n
  const int wm = (wid >> 2) * 64;
  const int wn = (wid & 3) * 32;
  v8f acc[4][2] = {};
  for (int k0 = 0; k0 < K; k0 += 32) {
    // A tile: row-major, coalesced global -> coalesced LDS
    const float4* ga = (const float4*)(A + (size_t)(m0 + arow) * K + k0 + aseg);
    float4* la = (float4*)(As + arow * LDA + aseg);
    la[0] = ga[0]; la[1] = ga[1];
    // B tile: coalesced global read along n, transposed scatter into LDS
    const v8h* gb = (const v8h*)(Bm + (size_t)(k0 + brow) * N + n0 + bseg);
    v8h t0 = gb[0], t1 = gb[1];
#pragma unroll
    for (int i = 0; i < 8; ++i) {
      BsT[(bseg + i) * LDBT + brow]     = t0[i];
      BsT[(bseg + 8 + i) * LDBT + brow] = t1[i];
    }
    if (k0 + 32 < K)
      __builtin_prefetch(A + (size_t)(m0 + arow) * K + k0 + 32 + aseg, 0, 0);
    __syncthreads();
    v16h b0 = fragBT(BsT, 0, wn, LDBT);
    v16h b1 = fragBT(BsT, 0, wn + 16, LDBT);
#pragma unroll
    for (int i = 0; i < 4; ++i) {
      v16h a = fragA(As, wm + i * 16, 0, LDA);
      acc[i][0] = wmma_f16(a, b0, acc[i][0]);
      acc[i][1] = wmma_f16(a, b1, acc[i][1]);
    }
    __syncthreads();
  }
#pragma unroll
  for (int i = 0; i < 4; ++i)
#pragma unroll
    for (int j = 0; j < 2; ++j) {
      const int mm = m0 + wm + i * 16;
      const int nn = n0 + wn + j * 16;
      if (half_out) storeCh(Ch, acc[i][j], mm, nn, N);
      else          storeC (Cf, acc[i][j], mm, nn, N);
    }
}

// ------------------------------------------------------------------
// K3: hedgehog softmax epilogue (one wave per (token, head)).
// ------------------------------------------------------------------
__global__ void k_hedgehog(const _Float16* __restrict__ y,
                           const float* __restrict__ bias,
                           _Float16* __restrict__ feat, float scale) {
  const int wid = threadIdx.x >> 5;
  const int ln  = threadIdx.x & 31;
  const int idx = blockIdx.x * 8 + wid;          // (b*T + t)*H + h
  const int bt  = idx >> 4;
  const int h   = idx & 15;
  const int b   = bt >> 11;
  const int t   = bt & (T_ - 1);
  const _Float16* yr = y + (size_t)bt * D_ + h * HD_;
  _Float16* fr = feat + (size_t)((b * H_ + h) * T_ + t) * F_;
  float y0 = (float)yr[2 * ln]     + bias[2 * ln];
  float y1 = (float)yr[2 * ln + 1] + bias[2 * ln + 1];
  float m = fmaxf(fabsf(y0), fabsf(y1));
#pragma unroll
  for (int off = 16; off > 0; off >>= 1) m = fmaxf(m, __shfl_xor(m, off));
  float e0 = __expf(y0 - m), e1 = __expf(y1 - m);
  float g0 = __expf(-y0 - m), g1 = __expf(-y1 - m);
  float s = e0 + e1 + g0 + g1;
#pragma unroll
  for (int off = 16; off > 0; off >>= 1) s += __shfl_xor(s, off);
  const float inv = scale / s;
  fr[2 * ln]           = (_Float16)(e0 * inv);
  fr[2 * ln + 1]       = (_Float16)(e1 * inv);
  fr[HD_ + 2 * ln]     = (_Float16)(g0 * inv);
  fr[HD_ + 2 * ln + 1] = (_Float16)(g1 * inv);
}

// ------------------------------------------------------------------
// K4: chunked causal linear attention.  One workgroup per (b,h), 8 waves.
// q/k tiles DMA'd with global_load_async_to_lds_b128 (ASYNCcnt).
// k also kept transposed (kst[f][c]) and v transposed (vst[d][c]) so the
// S-update A-frags and all v/state B-frags are contiguous b128 LDS loads.
// State kept TRANSPOSED: Stt[d][f] fp32.
// ------------------------------------------------------------------
__global__ void k_attn(const _Float16* __restrict__ qf,
                       const _Float16* __restrict__ kf,
                       const _Float16* __restrict__ vh,
                       _Float16* __restrict__ of) {
  constexpr int LDF  = 136;  // 128 + pad (f16), qs/ks rows
  constexpr int LDC  = 72;   // 64 + pad (f16), kst/vst/ss rows
  constexpr int LDS2 = 132;  // 128 + pad (fp32), Stt rows
  extern __shared__ char smem[];
  _Float16* qs  = (_Float16*)smem;            // [64][LDF]   q chunk (row-major)
  _Float16* ks  = qs  + C_ * LDF;             // [64][LDF]   k chunk (row-major)
  _Float16* kst = ks  + C_ * LDF;             // [128][LDC]  k chunk transposed [f][c]
  _Float16* vst = kst + F_ * LDC;             // [64][LDC]   v chunk transposed [d][c]
  _Float16* ss  = vst + HD_ * LDC;            // [64][LDC]   masked scores [c][m]
  float*    Stt = (float*)(ss + C_ * LDC);    // [64][LDS2]  state transposed [d][f]

  const int tid = threadIdx.x;
  const int wid = tid >> 5;
  const int b = blockIdx.x >> 4;
  const int h = blockIdx.x & 15;
  const _Float16* qbase = qf + (size_t)((b * H_ + h) * T_) * F_;
  const _Float16* kbase = kf + (size_t)((b * H_ + h) * T_) * F_;
  const _Float16* vbase = vh + (size_t)(b * T_) * D_ + h * HD_;
  _Float16*       obase = of + (size_t)(b * T_) * D_ + h * HD_;

  for (int i = tid; i < HD_ * LDS2; i += 256) Stt[i] = 0.f;
  __syncthreads();

  const int qrow = tid >> 2, qseg = (tid & 3) << 5;  // 64 rows x 32 f16 (q,k)
  const int vrow = tid >> 2, vseg = (tid & 3) << 4;  // 64 rows x 16 f16 (v)

  for (int ch = 0; ch < NCHUNK_; ++ch) {
    const int c0 = ch * C_;
    {
      // --- q,k row-major tiles: async DMA global -> LDS (4x16B each) ---
      const _Float16* gq = qbase + (size_t)(c0 + qrow) * F_ + qseg;
      const _Float16* gk = kbase + (size_t)(c0 + qrow) * F_ + qseg;
      _Float16* lq = qs + qrow * LDF + qseg;
      _Float16* lk = ks + qrow * LDF + qseg;
#pragma unroll
      for (int i = 0; i < 4; ++i) {
        async_b128(gq + 8 * i, lq + 8 * i);
        async_b128(gk + 8 * i, lk + 8 * i);
      }
      // --- k transposed copy (regular loads, L2-hot) ---
      const v8h* gk8 = (const v8h*)gk;
      v8h k0v = gk8[0], k1v = gk8[1], k2v = gk8[2], k3v = gk8[3];
#pragma unroll
      for (int i = 0; i < 8; ++i) {
        kst[(qseg + i)      * LDC + qrow] = k0v[i];
        kst[(qseg + 8 + i)  * LDC + qrow] = k1v[i];
        kst[(qseg + 16 + i) * LDC + qrow] = k2v[i];
        kst[(qseg + 24 + i) * LDC + qrow] = k3v[i];
      }
      // --- v transposed copy ---
      const v8h* gv8 = (const v8h*)(vbase + (size_t)(c0 + vrow) * D_ + vseg);
      v8h v0 = gv8[0], v1 = gv8[1];
#pragma unroll
      for (int i = 0; i < 8; ++i) {
        vst[(vseg + i)     * LDC + vrow] = v0[i];
        vst[(vseg + 8 + i) * LDC + vrow] = v1[i];
      }
      wait_async0();
    }
    __syncthreads();

    // --- intra-chunk scores = tril(qc @ kc^T) -> ss (f16) ---
#pragma unroll
    for (int s = 0; s < 2; ++s) {
      const int idx = wid * 2 + s;      // 16 tiles of 16x16
      const int ti = idx >> 2, tj = idx & 3;
      v8f acc = {};
#pragma unroll
      for (int kk = 0; kk < 4; ++kk) {
        v16h a  = fragA (qs, ti * 16, kk * 32, LDF);
        v16h bf = fragBT(ks, kk * 32, tj * 16, LDF);  // B=kc^T from row-major ks
        acc = wmma_f16(a, bf, acc);
      }
      const int ln = laneid();
      const int nc  = tj * 16 + (ln & 15);
      const int mrb = ti * 16 + ((ln >> 4) << 3);
      _Float16* pr = ss + mrb * LDC + nc;
#pragma unroll
      for (int j = 0; j < 8; ++j)
        pr[j * LDC] = (_Float16)((mrb + j >= nc) ? acc[j] : 0.0f);
    }
    __syncthreads();

    // --- o = qc @ S_excl + scores @ vc  (state read pre-update) ---
#pragma unroll
    for (int s = 0; s < 2; ++s) {
      const int idx = wid * 2 + s;
      const int ti = idx >> 2, tj = idx & 3;
      v8f acc = {};
#pragma unroll
      for (int kk = 0; kk < 4; ++kk) {
        v16h a  = fragA(qs, ti * 16, kk * 32, LDF);
        v16h bf = fragBTf32(Stt, kk * 32, tj * 16, LDS2);  // S[f][d] = Stt[d][f]
        acc = wmma_f16(a, bf, acc);
      }
#pragma unroll
      for (int kk = 0; kk < 2; ++kk) {
        v16h a  = fragA (ss, ti * 16, kk * 32, LDC);
        v16h bf = fragBT(vst, kk * 32, tj * 16, LDC);      // v[c][d] = vst[d][c]
        acc = wmma_f16(a, bf, acc);
      }
      storeCh(obase + (size_t)c0 * D_, acc, ti * 16, tj * 16, D_);
    }
    __syncthreads();

    // --- S += kc^T @ vc  (C/D on transposed state, contiguous b128) ---
#pragma unroll
    for (int s = 0; s < 4; ++s) {
      const int idx = wid * 4 + s;      // 32 tiles (8 f x 4 d)
      const int fi = idx >> 2, dj = idx & 3;
      v8f acc = loadCT(Stt, fi * 16, dj * 16, LDS2);
#pragma unroll
      for (int kk = 0; kk < 2; ++kk) {
        v16h a  = fragA (kst, fi * 16, kk * 32, LDC);  // A=kc^T from kst[f][c]
        v16h bf = fragBT(vst, kk * 32, dj * 16, LDC);
        acc = wmma_f16(a, bf, acc);
      }
      storeCT(Stt, acc, fi * 16, dj * 16, LDS2);
    }
    __syncthreads();
  }
}

// ------------------------------------------------------------------
extern "C" void kernel_launch(void* const* d_in, const int* in_sizes, int n_in,
                              void* d_out, int out_size, void* d_ws, size_t ws_size,
                              hipStream_t stream) {
  (void)in_sizes; (void)n_in; (void)out_size; (void)ws_size;
  const float* hs  = (const float*)d_in[0];
  const float* rs  = (const float*)d_in[1];
  const float* nw  = (const float*)d_in[2];
  const float* wq  = (const float*)d_in[3];
  const float* wk  = (const float*)d_in[4];
  const float* wv  = (const float*)d_in[5];
  const float* wo  = (const float*)d_in[6];
  const float* hqw = (const float*)d_in[7];
  const float* hqb = (const float*)d_in[8];
  const float* hkw = (const float*)d_in[9];
  const float* hkb = (const float*)d_in[10];

  float* out_o     = (float*)d_out;
  float* out_resid = out_o + (size_t)B_ * T_ * D_;

  char* ws = (char*)d_ws;
  const size_t MB = 1024 * 1024;
  _Float16* xh   = (_Float16*)(ws);             // 16 MB  x (f16)
  _Float16* wq_h = (_Float16*)(ws + 16 * MB);   //  2 MB folded wq
  _Float16* wk_h = (_Float16*)(ws + 18 * MB);   //  2 MB folded wk
  _Float16* wv_h = (_Float16*)(ws + 20 * MB);   //  2 MB
  _Float16* wo_h = (_Float16*)(ws + 22 * MB);   //  2 MB
  _Float16* yq   = (_Float16*)(ws + 24 * MB);   // 16 MB pre-softmax q
  _Float16* yk   = (_Float16*)(ws + 40 * MB);   // 16 MB pre-softmax k
  _Float16* vh   = (_Float16*)(ws + 56 * MB);   // 16 MB v projection
  _Float16* qfe  = (_Float16*)(ws + 72 * MB);   // 32 MB q features [B,H,T,F]
  _Float16* kfe  = (_Float16*)(ws + 104 * MB);  // 32 MB k features
  _Float16* ofe  = (_Float16*)(ws + 136 * MB);  // 16 MB attention out (f16)

  k_addnorm<<<B_ * T_, 256, 0, stream>>>(hs, rs, nw, out_resid, xh);
  k_fold<<<4096, 256, 0, stream>>>(wq, hqw, wq_h);
  k_fold<<<4096, 256, 0, stream>>>(wk, hkw, wk_h);
  k_cvt <<<4096, 256, 0, stream>>>(wv, wv_h);
  k_cvt <<<4096, 256, 0, stream>>>(wo, wo_h);

  dim3 gg(D_ / 128, (B_ * T_) / 128);  // (8, 64)
  k_gemm<<<gg, 256, 0, stream>>>(xh, wq_h, nullptr, yq, B_ * T_, D_, D_, 1);
  k_gemm<<<gg, 256, 0, stream>>>(xh, wk_h, nullptr, yk, B_ * T_, D_, D_, 1);
  k_gemm<<<gg, 256, 0, stream>>>(xh, wv_h, nullptr, vh, B_ * T_, D_, D_, 1);

  k_hedgehog<<<(B_ * T_ * H_) / 8, 256, 0, stream>>>(yq, hqb, qfe,
                                                     0.08838834764831845f); // F^-0.5
  k_hedgehog<<<(B_ * T_ * H_) / 8, 256, 0, stream>>>(yk, hkb, kfe, 1.0f);

  const size_t smem_attn =
      (size_t)(2 * C_ * 136 + (F_ + HD_ + C_) * 72) * sizeof(_Float16)
      + (size_t)(HD_ * 132) * sizeof(float);  // 105,472 B
  k_attn<<<B_ * H_, 256, smem_attn, stream>>>(qfe, kfe, vh, ofe);

  k_gemm<<<gg, 256, 0, stream>>>(ofe, wo_h, out_o, nullptr, B_ * T_, D_, D_, 0);
}
